// DeepSCTransformerBlock_87093346828520
// MI455X (gfx1250) — compile-verified, matrix operand order checked by reading, not run on hardware
//
#include <hip/hip_runtime.h>

// ---------------------------------------------------------------------------
// DeepSC transformer block for MI455X (gfx1250, wave32, WMMA bf16 16x16x32,
// async global->LDS double-buffered GEMM pipeline)
// ---------------------------------------------------------------------------

typedef __attribute__((ext_vector_type(16))) __bf16 bf16x16;
typedef __attribute__((ext_vector_type(8)))  float  f32x8;

__device__ __forceinline__ unsigned short f2bf(float f) {
  union { float f; unsigned int u; } v; v.f = f;
  unsigned int u = v.u;
  unsigned int r = u + 0x7FFFu + ((u >> 16) & 1u);   // round-to-nearest-even
  return (unsigned short)(r >> 16);
}

// CDNA5 async copy: global -> LDS, 16B per lane, tracked by ASYNCcnt.
// LDS address = low 32 bits of the generic pointer (flat LDS aperture mapping).
__device__ __forceinline__ void async_ld_b128(unsigned short* lds, const unsigned short* g) {
  unsigned lds_off = (unsigned)(unsigned long long)lds;
  asm volatile("global_load_async_to_lds_b128 %0, %1, off"
               :: "v"(lds_off), "v"(g) : "memory");
}
__device__ __forceinline__ void wait_asynccnt0() {
  asm volatile("s_wait_asynccnt 0x0" ::: "memory");
}
__device__ __forceinline__ void wait_dscnt0() {
  asm volatile("s_wait_dscnt 0x0" ::: "memory");
}

// ---------------------------------------------------------------------------
// Generic WMMA GEMM:  C[M][N] = A[M][K](bf16) * Bt[N][K](bf16)^T  (+bias)(+epi)(+res)
// Block tile 128x64, 8 waves, each wave 32x32 (2x2 WMMA 16x16 tiles), BK=32.
// Double-buffered LDS filled by GLOBAL_LOAD_ASYNC_TO_LDS_B128 so the next
// K-tile's transfer overlaps the current tile's WMMAs.
// Batched over blockIdx.z with (zb, zh) = (z / Hdim, z % Hdim) strides so the
// per-(batch, head) attention slices address correctly.
// EPI: 0 = none, 1 = exact GELU.  RES: add fp32 residual.  OUTF32: fp32 vs bf16 out.
// ---------------------------------------------------------------------------
template<int EPI, bool RES, bool OUTF32>
__global__ __launch_bounds__(256) void wmma_gemm(
    const unsigned short* __restrict__ A,  long lda, long sAb, long sAh,
    const unsigned short* __restrict__ Bt, long ldb, long sBb, long sBh,
    void* __restrict__ Cv, long ldc, long sCb, long sCh,
    const float* __restrict__ bias,
    const float* __restrict__ res, long ldr, long sRb, long sRh,
    int M, int N, int K, int Hdim)
{
  __shared__ unsigned short As[2][128 * 32];
  __shared__ unsigned short Bs[2][64 * 32];

  const int bz = blockIdx.z;
  const int zb = bz / Hdim, zh = bz % Hdim;
  A  += zb * sAb + zh * sAh;
  Bt += zb * sBb + zh * sBh;
  const long coff = (long)zb * sCb + (long)zh * sCh;
  const long roff = (long)zb * sRb + (long)zh * sRh;

  const int tid  = threadIdx.x;
  const int lane = tid & 31;
  const int wave = tid >> 5;
  const int wr   = wave >> 1;     // 0..3  -> 32-row strip
  const int wc   = wave & 1;      // 0..1  -> 32-col strip
  const long row0 = (long)blockIdx.y * 128;
  const long col0 = (long)blockIdx.x * 64;

  f32x8 acc[2][2] = {};

  const int ar = tid >> 1, aks = (tid & 1) * 16;   // A tile: 128 rows x 32k
  const int br = tid >> 2, bks = (tid & 3) * 8;    // B tile:  64 rows x 32k

  // issue the async fill of one K-tile into buffer p
  auto issue_tile = [&](int k0, int p) {
    const unsigned short* ga = A + (row0 + ar) * lda + k0 + aks;
    async_ld_b128(&As[p][ar * 32 + aks],     ga);
    async_ld_b128(&As[p][ar * 32 + aks + 8], ga + 8);
    async_ld_b128(&Bs[p][br * 32 + bks], Bt + (col0 + br) * ldb + k0 + bks);
  };

  // prologue: fill buffer 0
  issue_tile(0, 0);
  wait_asynccnt0();
  __syncthreads();

  int p = 0;
  for (int k0 = 0; k0 < K; k0 += 32, p ^= 1) {
    if (k0 + 32 < K) issue_tile(k0 + 32, p ^ 1);   // overlaps with WMMAs below

    const int kofs = (lane >> 4) * 16;   // lanes 16-31 hold K=16..31 half
    const int lrow = lane & 15;
    bf16x16 a0 = *(const bf16x16*)&As[p][(wr * 32 +      lrow) * 32 + kofs];
    bf16x16 a1 = *(const bf16x16*)&As[p][(wr * 32 + 16 + lrow) * 32 + kofs];
    bf16x16 b0 = *(const bf16x16*)&Bs[p][(wc * 32 +      lrow) * 32 + kofs];
    bf16x16 b1 = *(const bf16x16*)&Bs[p][(wc * 32 + 16 + lrow) * 32 + kofs];

    acc[0][0] = __builtin_amdgcn_wmma_f32_16x16x32_bf16(false, a0, false, b0, (short)0, acc[0][0], false, false);
    acc[0][1] = __builtin_amdgcn_wmma_f32_16x16x32_bf16(false, a0, false, b1, (short)0, acc[0][1], false, false);
    acc[1][0] = __builtin_amdgcn_wmma_f32_16x16x32_bf16(false, a1, false, b0, (short)0, acc[1][0], false, false);
    acc[1][1] = __builtin_amdgcn_wmma_f32_16x16x32_bf16(false, a1, false, b1, (short)0, acc[1][1], false, false);

    wait_dscnt0();     // my fragment reads of buf[p] retired
    wait_asynccnt0();  // my async fill of buf[p^1] landed in LDS
    __syncthreads();   // all waves agree before buffers swap roles
  }

  // 16x16 f32 C/D layout: VGPR r, lane l -> row = r + (l>>4)*8, col = l&15
  const int lrow = lane & 15;
  const int lhi  = (lane >> 4) * 8;
  for (int mi = 0; mi < 2; ++mi) {
    for (int ni = 0; ni < 2; ++ni) {
#pragma unroll
      for (int r = 0; r < 8; ++r) {
        long gm = row0 + wr * 32 + mi * 16 + lhi + r;
        long gn = col0 + wc * 32 + ni * 16 + lrow;
        float v = acc[mi][ni][r];
        if (bias) v += bias[gn];
        if (EPI == 1) v = 0.5f * v * (1.0f + erff(v * 0.70710678118654752f));
        if (RES) v += res[roff + gm * ldr + gn];
        if (OUTF32) ((float*)Cv)[coff + gm * ldc + gn] = v;
        else        ((unsigned short*)Cv)[coff + gm * ldc + gn] = f2bf(v);
      }
    }
  }
}

// ---------------------------------------------------------------------------
// LayerNorm over last dim D (fp32 in, bf16 out). One block per row.
// ---------------------------------------------------------------------------
__global__ __launch_bounds__(256) void ln_kernel(
    const float* __restrict__ x, const float* __restrict__ g,
    const float* __restrict__ bta, unsigned short* __restrict__ y, int D)
{
  __shared__ float red[256];
  const long row = blockIdx.x;
  const float* xr = x + row * D;
  const int t = threadIdx.x;

  float s = 0.f;
  for (int i = t; i < D; i += 256) s += xr[i];
  red[t] = s; __syncthreads();
  for (int st = 128; st > 0; st >>= 1) { if (t < st) red[t] += red[t + st]; __syncthreads(); }
  const float mean = red[0] / (float)D;
  __syncthreads();

  float s2 = 0.f;
  for (int i = t; i < D; i += 256) { float d = xr[i] - mean; s2 += d * d; }
  red[t] = s2; __syncthreads();
  for (int st = 128; st > 0; st >>= 1) { if (t < st) red[t] += red[t + st]; __syncthreads(); }
  const float rstd = rsqrtf(red[0] / (float)D + 1e-5f);

  unsigned short* yr = y + row * D;
  for (int i = t; i < D; i += 256)
    yr[i] = f2bf((xr[i] - mean) * rstd * g[i] + bta[i]);
}

// ---------------------------------------------------------------------------
// softmax(S*scale) * mask  [optionally renormalized by sum(|.|)+1e-8] -> bf16
// One block per (bh, row). G == 1024.
// ---------------------------------------------------------------------------
__global__ __launch_bounds__(256) void softmax_mask_kernel(
    const float* __restrict__ S, const float* __restrict__ Mask,
    unsigned short* __restrict__ P, int G, int H, float scale, int renorm)
{
  __shared__ float e[1024];
  __shared__ float red[256];
  const int bh = blockIdx.y;
  const int b  = bh / H;
  const long row = blockIdx.x;
  const float* srow = S + ((long)bh * G + row) * G;
  const float* mrow = Mask + ((long)b * G + row) * G;
  unsigned short* prow = P + ((long)bh * G + row) * G;
  const int t = threadIdx.x;

  float mx = -3.0e38f;
  for (int j = t; j < G; j += 256) { float v = srow[j] * scale; e[j] = v; mx = fmaxf(mx, v); }
  red[t] = mx; __syncthreads();
  for (int st = 128; st > 0; st >>= 1) { if (t < st) red[t] = fmaxf(red[t], red[t + st]); __syncthreads(); }
  mx = red[0]; __syncthreads();

  float sum = 0.f;
  for (int j = t; j < G; j += 256) { float v = expf(e[j] - mx); e[j] = v; sum += v; }
  red[t] = sum; __syncthreads();
  for (int st = 128; st > 0; st >>= 1) { if (t < st) red[t] += red[t + st]; __syncthreads(); }
  const float inv = 1.0f / red[0]; __syncthreads();

  float asum = 0.f;
  for (int j = t; j < G; j += 256) { float v = e[j] * inv * mrow[j]; e[j] = v; asum += fabsf(v); }
  if (renorm) {
    red[t] = asum; __syncthreads();
    for (int st = 128; st > 0; st >>= 1) { if (t < st) red[t] += red[t + st]; __syncthreads(); }
    const float d = 1.0f / (red[0] + 1e-8f); __syncthreads();
    for (int j = t; j < G; j += 256) prow[j] = f2bf(e[j] * d);
  } else {
    for (int j = t; j < G; j += 256) prow[j] = f2bf(e[j]);
  }
}

// V[bg][H*HD] -> Vt[(b*H+h)*HD + d][G]  (per-head transpose, bf16)
__global__ __launch_bounds__(256) void transpose_heads_kernel(
    const unsigned short* __restrict__ V, unsigned short* __restrict__ Vt,
    int G, int H, int HD)
{
  long idx = (long)blockIdx.x * 256 + threadIdx.x;
  int d = (int)(idx % HD); long r = idx / HD;
  int h = (int)(r % H); r /= H;
  int g = (int)(r % G); int b = (int)(r / G);
  Vt[(((long)b * H + h) * HD + d) * G + g] = V[(((long)b * G + g) * H + h) * HD + d];
}

// per-head concat along HD: Cat[bg][h*2HD + c] = c<HD ? Xg[bg][h*HD+c] : Xe[bg][h*HD+c-HD]
__global__ __launch_bounds__(256) void concat_heads_kernel(
    const unsigned short* __restrict__ Xg, const unsigned short* __restrict__ Xe,
    unsigned short* __restrict__ Cat, int G, int H, int HD)
{
  long idx = (long)blockIdx.x * 256 + threadIdx.x;
  int c = (int)(idx % (2 * HD)); long r = idx / (2 * HD);
  int h = (int)(r % H); r /= H;
  int g = (int)(r % G); int b = (int)(r / G);
  unsigned short v = (c < HD)
      ? Xg[(((long)b * G + g) * H + h) * HD + c]
      : Xe[(((long)b * G + g) * H + h) * HD + (c - HD)];
  Cat[(((long)b * G + g) * H + h) * (2 * HD) + c] = v;
}

// W[K][N] fp32  ->  Wt[N][K] bf16  (GEMM wants B^T row-major)
__global__ __launch_bounds__(256) void cvt_t_kernel(
    const float* __restrict__ W, unsigned short* __restrict__ Wt, int K, int N)
{
  long idx = (long)blockIdx.x * 256 + threadIdx.x;
  if (idx >= (long)K * N) return;
  int n = (int)(idx % N); long k = idx / N;
  Wt[(long)n * K + k] = f2bf(W[idx]);
}

// ---------------------------------------------------------------------------
extern "C" void kernel_launch(void* const* d_in, const int* in_sizes, int n_in,
                              void* d_out, int out_size, void* d_ws, size_t ws_size,
                              hipStream_t stream)
{
  (void)in_sizes; (void)n_in; (void)out_size; (void)ws_size;
  constexpr int Bn = 4, G = 1024, D = 512, H = 8, HD = 64;
  constexpr int M = Bn * G;        // 4096 token rows
  constexpr int Dff = 4 * D;       // 2048
  constexpr long GG = (long)G * G;
  const float SCALE = 0.125f;      // 1/sqrt(HD)

  const float* gene_emb = (const float*)d_in[0];
  const float* expr_emb = (const float*)d_in[1];
  const float* Mask     = (const float*)d_in[2];
  const float* gene_wq = (const float*)d_in[3];  const float* gene_bq = (const float*)d_in[4];
  const float* gene_wk = (const float*)d_in[5];  const float* gene_bk = (const float*)d_in[6];
  const float* gene_wv = (const float*)d_in[7];  const float* gene_bv = (const float*)d_in[8];
  const float* expr_wq = (const float*)d_in[9];  const float* expr_bq = (const float*)d_in[10];
  const float* expr_wk = (const float*)d_in[11]; const float* expr_bk = (const float*)d_in[12];
  const float* expr_wv = (const float*)d_in[13]; const float* expr_bv = (const float*)d_in[14];
  const float* fused_wq = (const float*)d_in[15]; const float* fused_bq = (const float*)d_in[16];
  const float* fused_wk = (const float*)d_in[17]; const float* fused_bk = (const float*)d_in[18];
  const float* gout_w = (const float*)d_in[19]; const float* gout_b = (const float*)d_in[20];
  const float* eout_w = (const float*)d_in[21]; const float* eout_b = (const float*)d_in[22];
  const float* ln_g1_g = (const float*)d_in[23]; const float* ln_g1_b = (const float*)d_in[24];
  const float* ln_g2_g = (const float*)d_in[25]; const float* ln_g2_b = (const float*)d_in[26];
  const float* ln_e1_g = (const float*)d_in[27]; const float* ln_e1_b = (const float*)d_in[28];
  const float* ln_e2_g = (const float*)d_in[29]; const float* ln_e2_b = (const float*)d_in[30];
  const float* ffn_g_w1 = (const float*)d_in[31]; const float* ffn_g_b1 = (const float*)d_in[32];
  const float* ffn_g_w2 = (const float*)d_in[33]; const float* ffn_g_b2 = (const float*)d_in[34];
  const float* ffn_e_w1 = (const float*)d_in[35]; const float* ffn_e_b1 = (const float*)d_in[36];
  const float* ffn_e_w2 = (const float*)d_in[37]; const float* ffn_e_b2 = (const float*)d_in[38];

  float* outg = (float*)d_out;
  float* oute = outg + (long)M * D;

  // ---- workspace carve-up ----
  char* wsb = (char*)d_ws;
  size_t off = 0;
  auto aU16 = [&](long n) { unsigned short* p = (unsigned short*)(wsb + off);
                            off = (off + (size_t)n * 2 + 255) & ~(size_t)255; return p; };
  auto aF32 = [&](long n) { float* p = (float*)(wsb + off);
                            off = (off + (size_t)n * 4 + 255) & ~(size_t)255; return p; };

  unsigned short *wt_gq = aU16((long)D*D), *wt_gk = aU16((long)D*D), *wt_gv = aU16((long)D*D);
  unsigned short *wt_eq = aU16((long)D*D), *wt_ek = aU16((long)D*D), *wt_ev = aU16((long)D*D);
  unsigned short *wt_go = aU16((long)D*D), *wt_eo = aU16((long)D*D);
  unsigned short *wt_g1 = aU16((long)D*Dff), *wt_g2 = aU16((long)D*Dff);
  unsigned short *wt_e1 = aU16((long)D*Dff), *wt_e2 = aU16((long)D*Dff);
  unsigned short *wt_fq = aU16(2L*HD*HD), *wt_fk = aU16(2L*HD*HD);
  unsigned short *xa = aU16((long)M*D), *xb = aU16((long)M*D), *x2 = aU16((long)M*D);
  unsigned short *Qb = aU16((long)M*D), *Kb = aU16((long)M*D), *Vb = aU16((long)M*D);
  unsigned short *Q2 = aU16((long)M*D), *K2 = aU16((long)M*D), *V2 = aU16((long)M*D);
  unsigned short *Vt = aU16((long)M*D), *att = aU16((long)M*D);
  unsigned short *hid = aU16((long)M*Dff);
  unsigned short *Qcat = aU16((long)M*2*D), *Kcat = aU16((long)M*2*D);
  unsigned short *probs = aU16(32L*GG);
  float *scores = aF32(32L*GG);
  float *hresg = aF32((long)M*D), *hrese = aF32((long)M*D);

  // ---- helpers ----
  auto cvt = [&](const float* W, unsigned short* Wt, int K, int N) {
    long tot = (long)K * N;
    cvt_t_kernel<<<dim3((unsigned)((tot + 255) / 256)), 256, 0, stream>>>(W, Wt, K, N);
  };
  auto gemm_b16 = [&](const unsigned short* A, const unsigned short* Bt,
                      unsigned short* C, const float* bias, int Mm, int Nn, int Kk) {
    wmma_gemm<0, false, false><<<dim3(Nn/64, Mm/128, 1), 256, 0, stream>>>(
        A, Kk, 0, 0, Bt, Kk, 0, 0, (void*)C, Nn, 0, 0, bias, nullptr, 0, 0, 0, Mm, Nn, Kk, 1);
  };
  auto gemm_gelu = [&](const unsigned short* A, const unsigned short* Bt,
                       unsigned short* C, const float* bias, int Mm, int Nn, int Kk) {
    wmma_gemm<1, false, false><<<dim3(Nn/64, Mm/128, 1), 256, 0, stream>>>(
        A, Kk, 0, 0, Bt, Kk, 0, 0, (void*)C, Nn, 0, 0, bias, nullptr, 0, 0, 0, Mm, Nn, Kk, 1);
  };
  auto gemm_res = [&](const unsigned short* A, const unsigned short* Bt,
                      float* C, const float* bias, const float* res,
                      int Mm, int Nn, int Kk) {
    wmma_gemm<0, true, true><<<dim3(Nn/64, Mm/128, 1), 256, 0, stream>>>(
        A, Kk, 0, 0, Bt, Kk, 0, 0, (void*)C, Nn, 0, 0, bias, res, Nn, 0, 0, Mm, Nn, Kk, 1);
  };
  auto attn_scores = [&](const unsigned short* Q, const unsigned short* Kmat) {
    wmma_gemm<0, false, true><<<dim3(G/64, G/128, Bn*H), 256, 0, stream>>>(
        Q, D, (long)G*D, HD, Kmat, D, (long)G*D, HD,
        (void*)scores, G, (long)H*GG, GG,
        nullptr, nullptr, 0, 0, 0, G, G, HD, H);
  };
  auto attn_av = [&]() {
    wmma_gemm<0, false, false><<<dim3(HD/64, G/128, Bn*H), 256, 0, stream>>>(
        probs, G, (long)H*GG, GG, Vt, G, (long)H*HD*G, (long)HD*G,
        (void*)att, D, (long)G*D, HD,
        nullptr, nullptr, 0, 0, 0, G, HD, G, H);
  };

  // ---- weight conversion + transpose (fp32 -> bf16, B^T layout) ----
  cvt(gene_wq, wt_gq, D, D);  cvt(gene_wk, wt_gk, D, D);  cvt(gene_wv, wt_gv, D, D);
  cvt(expr_wq, wt_eq, D, D);  cvt(expr_wk, wt_ek, D, D);  cvt(expr_wv, wt_ev, D, D);
  cvt(gout_w, wt_go, D, D);   cvt(eout_w, wt_eo, D, D);
  cvt(ffn_g_w1, wt_g1, D, Dff); cvt(ffn_g_w2, wt_g2, Dff, D);
  cvt(ffn_e_w1, wt_e1, D, Dff); cvt(ffn_e_w2, wt_e2, Dff, D);
  cvt(fused_wq, wt_fq, 2*HD, HD); cvt(fused_wk, wt_fk, 2*HD, HD);

  // ================= Gene branch =================
  ln_kernel<<<M, 256, 0, stream>>>(gene_emb, ln_g1_g, ln_g1_b, xa, D);
  gemm_b16(xa, wt_gq, Qb, gene_bq, M, D, D);
  gemm_b16(xa, wt_gk, Kb, gene_bk, M, D, D);
  gemm_b16(xa, wt_gv, Vb, gene_bv, M, D, D);
  attn_scores(Qb, Kb);
  softmax_mask_kernel<<<dim3(G, Bn*H), 256, 0, stream>>>(scores, Mask, probs, G, H, SCALE, 1);
  transpose_heads_kernel<<<(M * D) / 256, 256, 0, stream>>>(Vb, Vt, G, H, HD);
  attn_av();
  gemm_res(att, wt_go, hresg, gout_b, gene_emb, M, D, D);            // h_gene
  ln_kernel<<<M, 256, 0, stream>>>(hresg, ln_g2_g, ln_g2_b, x2, D);
  gemm_gelu(x2, wt_g1, hid, ffn_g_b1, M, Dff, D);
  gemm_res(hid, wt_g2, outg, ffn_g_b2, hresg, M, D, Dff);            // out_gene

  // ================= Expr branch =================
  ln_kernel<<<M, 256, 0, stream>>>(gene_emb, ln_e1_g, ln_e1_b, xa, D);
  ln_kernel<<<M, 256, 0, stream>>>(expr_emb, ln_e1_g, ln_e1_b, xb, D);
  gemm_b16(xa, wt_gq, Qb, gene_bq, M, D, D);   // Qg
  gemm_b16(xa, wt_gk, Kb, gene_bk, M, D, D);   // Kg
  gemm_b16(xb, wt_eq, Q2, expr_bq, M, D, D);   // Qe
  gemm_b16(xb, wt_ek, K2, expr_bk, M, D, D);   // Ke
  gemm_b16(xb, wt_ev, V2, expr_bv, M, D, D);   // Ve
  concat_heads_kernel<<<(M * 2 * D) / 256, 256, 0, stream>>>(Qb, Q2, Qcat, G, H, HD);
  concat_heads_kernel<<<(M * 2 * D) / 256, 256, 0, stream>>>(Kb, K2, Kcat, G, H, HD);
  // fused per-head projections (weight shared across batch: strides 0)
  wmma_gemm<0, false, false><<<dim3(1, G/128, Bn*H), 256, 0, stream>>>(
      Qcat, 2*D, (long)G*2*D, 2*HD, wt_fq, 2*HD, 0, 0,
      (void*)Qb, D, (long)G*D, HD,
      fused_bq, nullptr, 0, 0, 0, G, HD, 2*HD, H);
  wmma_gemm<0, false, false><<<dim3(1, G/128, Bn*H), 256, 0, stream>>>(
      Kcat, 2*D, (long)G*2*D, 2*HD, wt_fk, 2*HD, 0, 0,
      (void*)Kb, D, (long)G*D, HD,
      fused_bk, nullptr, 0, 0, 0, G, HD, 2*HD, H);
  attn_scores(Qb, Kb);
  softmax_mask_kernel<<<dim3(G, Bn*H), 256, 0, stream>>>(scores, Mask, probs, G, H, SCALE, 0);
  transpose_heads_kernel<<<(M * D) / 256, 256, 0, stream>>>(V2, Vt, G, H, HD);
  attn_av();
  gemm_res(att, wt_eo, hrese, eout_b, expr_emb, M, D, D);            // h_expr
  ln_kernel<<<M, 256, 0, stream>>>(hrese, ln_e2_g, ln_e2_b, x2, D);
  gemm_gelu(x2, wt_e1, hid, ffn_e_b1, M, Dff, D);
  gemm_res(hid, wt_e2, oute, ffn_e_b2, hrese, M, D, Dff);            // out_expr
}
